// GRN_67216238182732
// MI455X (gfx1250) — matrix operand order repaired and checked by reading, not compiled, not run on hardware
//
#include <hip/hip_runtime.h>

#define B_ 8
#define L_ 512
#define D_ 64
#define S_ 8

typedef __attribute__((ext_vector_type(16))) __bf16 v16bf;
typedef __attribute__((ext_vector_type(8)))  float  v8f;

__device__ __forceinline__ v8f wmma_bf16(v16bf a, v16bf b, v8f c) {
  // 8 args: (neg_a, A, neg_b, B, c_mod, C, reuse_a, reuse_b)
  return __builtin_amdgcn_wmma_f32_16x16x32_bf16(false, a, false, b, (short)0, c,
                                                 false, false);
}

__device__ __forceinline__ float sigm(float x) { return 1.0f / (1.0f + __expf(-x)); }

// ---------------------------------------------------------------------------
// Kernel 1: A[b,s] = lstm1[b] @ W[s]  (fp32), split to bf16 hi/lo, pre-packed
// into the CDNA5 wave32 WMMA 16x32 bf16 A-fragment layout.
//   A-frag: lane = m + 16*((k%32)>>3 & 1), elem e = (k%32 & 7) + 8*((k%32)>=16)
// Packed index: [((b*S+s)*32 + itile)*2 + kstep][lane][e]   (16 bf16 per lane)
// ---------------------------------------------------------------------------
__global__ __launch_bounds__(256) void grn_prep_A(const float* __restrict__ lstm1,
                                                  const float* __restrict__ W,
                                                  __bf16* __restrict__ Ahi,
                                                  __bf16* __restrict__ Alo) {
  const int t   = threadIdx.x;
  const int dp  = t & 63;          // output column e' of A
  const int il  = t >> 6;          // 0..3 rows per block
  const int blk = blockIdx.x;      // B*S*(L/4) = 8192 blocks
  const int i4  = blk & 127;
  const int s   = (blk >> 7) & 7;
  const int b   = blk >> 10;
  const int i   = i4 * 4 + il;

  const float* x = lstm1 + ((size_t)b * L_ + i) * D_;
  const float* w = W + (size_t)s * D_ * D_ + dp;
  float acc = 0.f;
#pragma unroll
  for (int d = 0; d < D_; ++d) acc = fmaf(x[d], w[(size_t)d * D_], acc);

  const __bf16 hi = (__bf16)acc;
  const __bf16 lo = (__bf16)(acc - (float)hi);

  const int itile = i >> 4, m = i & 15;
  const int kstep = dp >> 5, kk = dp & 31;
  const int lane  = m + (((kk >> 3) & 1) << 4);
  const int e     = (kk & 7) + ((kk >> 4) << 3);
  const size_t idx =
      ((((size_t)(b * S_ + s) * 32 + itile) * 2 + kstep) * 32 + lane) * 16 + e;
  Ahi[idx] = hi;
  Alo[idx] = lo;
}

// ---------------------------------------------------------------------------
// Kernel 2: pack lstm2 (bf16 hi/lo) into the 32x16 bf16 B-fragment layout.
//   B-frag: lane = n + 16*((k%32)>>3 & 1), elem e = (k%32 & 7) + 8*((k%32)>=16)
// ---------------------------------------------------------------------------
__global__ __launch_bounds__(256) void grn_pack_B(const float* __restrict__ lstm2,
                                                  __bf16* __restrict__ Bhi,
                                                  __bf16* __restrict__ Blo) {
  const int gid = blockIdx.x * 256 + threadIdx.x;  // B*L*D = 262144 threads
  const int d = gid & 63;
  const int j = (gid >> 6) & 511;
  const int b = gid >> 15;

  const float x = lstm2[gid];
  const __bf16 hi = (__bf16)x;
  const __bf16 lo = (__bf16)(x - (float)hi);

  const int jt = j >> 4, n = j & 15;
  const int kstep = d >> 5, kk = d & 31;
  const int lane = n + (((kk >> 3) & 1) << 4);
  const int e    = (kk & 7) + ((kk >> 4) << 3);
  const size_t idx = (((size_t)(b * 32 + jt) * 2 + kstep) * 32 + lane) * 16 + e;
  Bhi[idx] = hi;
  Blo[idx] = lo;
}

// ---------------------------------------------------------------------------
// Kernel 3: v1/v2/g1/g2 projections, (B, L, S) fp32, S contiguous.
// ---------------------------------------------------------------------------
__global__ __launch_bounds__(256) void grn_prep_vec(const float* __restrict__ lstm1,
                                                    const float* __restrict__ lstm2,
                                                    const float* __restrict__ V,
                                                    const float* __restrict__ Wg,
                                                    float* __restrict__ v1,
                                                    float* __restrict__ v2,
                                                    float* __restrict__ g1,
                                                    float* __restrict__ g2) {
  const int gid = blockIdx.x * 256 + threadIdx.x;  // 32768 threads
  const int s = gid & 7;
  const int i = (gid >> 3) & 511;
  const int b = gid >> 12;

  const float* x1 = lstm1 + ((size_t)b * L_ + i) * D_;
  const float* x2 = lstm2 + ((size_t)b * L_ + i) * D_;
  const float* vr = V  + (size_t)s * 2 * D_;
  const float* gr = Wg + (size_t)s * 2 * D_;
  float a1 = 0.f, a2 = 0.f, a3 = 0.f, a4 = 0.f;
#pragma unroll
  for (int d = 0; d < D_; ++d) {
    a1 = fmaf(x1[d], vr[d],      a1);
    a2 = fmaf(x2[d], vr[D_ + d], a2);
    a3 = fmaf(x1[d], gr[d],      a3);
    a4 = fmaf(x2[d], gr[D_ + d], a4);
  }
  v1[gid] = a1;
  v2[gid] = a2;
  g1[gid] = a3;
  g2[gid] = a4;
}

// ---------------------------------------------------------------------------
// Kernel 4: main fused WMMA kernel. One wave per 16x16 output tile.
//   R_w(16x16, 8 slots) via split-bf16 WMMA (hi*hi + hi*lo + lo*hi),
//   then gate/sigmoid/bias/u-reduction epilogue, coalesced store.
// C-frag mapping: lane l, reg r -> M = r + 8*(l>=16), N = l%16.
// ---------------------------------------------------------------------------
__global__ __launch_bounds__(256) void grn_main(
    const __bf16* __restrict__ Ahi, const __bf16* __restrict__ Alo,
    const __bf16* __restrict__ Bhi, const __bf16* __restrict__ Blo,
    const float* __restrict__ v1, const float* __restrict__ v2,
    const float* __restrict__ g1, const float* __restrict__ g2,
    const float* __restrict__ bias, const float* __restrict__ bg,
    const float* __restrict__ u, float* __restrict__ out) {
  const int lane = threadIdx.x & 31;
  const int wave = threadIdx.x >> 5;
  const int b    = blockIdx.x >> 7;                  // 128 blocks per batch
  const int wg   = (blockIdx.x & 127) * 8 + wave;    // 0..1023 tiles in batch
  const int it   = wg >> 5;
  const int jt   = wg & 31;

  // B fragments (shared across all 8 slots), 2 k-steps, hi and lo parts.
  const __bf16* pbh = Bhi + (((size_t)(b * 32 + jt) * 2) * 32 + lane) * 16;
  const __bf16* pbl = Blo + (((size_t)(b * 32 + jt) * 2) * 32 + lane) * 16;
  const v16bf bh0 = *(const v16bf*)(pbh);
  const v16bf bh1 = *(const v16bf*)(pbh + 512);
  const v16bf bl0 = *(const v16bf*)(pbl);
  const v16bf bl1 = *(const v16bf*)(pbl + 512);

  const size_t abase = ((((size_t)(b * S_) * 32 + it) * 2) * 32 + lane) * 16;
  v8f acc[S_];
#pragma unroll
  for (int s = 0; s < S_; ++s) {
    const __bf16* pah = Ahi + abase + (size_t)s * 32768;
    const __bf16* pal = Alo + abase + (size_t)s * 32768;
    __builtin_prefetch(pah + 32768, 0, 1);  // next slot's A-hi fragments
    const v16bf ah0 = *(const v16bf*)(pah);
    const v16bf ah1 = *(const v16bf*)(pah + 512);
    const v16bf al0 = *(const v16bf*)(pal);
    const v16bf al1 = *(const v16bf*)(pal + 512);
    v8f c = {};
    c = wmma_bf16(ah0, bh0, c);   // hi * hi   (K = 0..31)
    c = wmma_bf16(ah1, bh1, c);   // hi * hi   (K = 32..63)
    c = wmma_bf16(ah0, bl0, c);   // hi * lo
    c = wmma_bf16(ah1, bl1, c);
    c = wmma_bf16(al0, bh0, c);   // lo * hi
    c = wmma_bf16(al1, bh1, c);
    acc[s] = c;
  }

  // Fused epilogue.
  const int n = lane & 15;
  const int j = jt * 16 + n;
  const v8f v2r = *(const v8f*)(v2 + ((size_t)b * L_ + j) * S_);
  const v8f g2r = *(const v8f*)(g2 + ((size_t)b * L_ + j) * S_);
  float bb[S_], bgr[S_], ur[S_];
#pragma unroll
  for (int s = 0; s < S_; ++s) {
    bb[s]  = bias[s];
    bgr[s] = bg[s];
    ur[s]  = u[s];
  }
  const int mbase = (lane >> 4) << 3;  // 0 or 8
#pragma unroll
  for (int r = 0; r < 8; ++r) {
    const int i = it * 16 + mbase + r;
    const v8f v1r = *(const v8f*)(v1 + ((size_t)b * L_ + i) * S_);
    const v8f g1r = *(const v8f*)(g1 + ((size_t)b * L_ + i) * S_);
    float sum = 0.f;
#pragma unroll
    for (int s = 0; s < S_; ++s) {
      const float gate = sigm(g1r[s] + g2r[s] + bgr[s]);
      const float rv   = sigm(v1r[s] + v2r[s]);
      const float val  = fmaf(gate, acc[s][r], fmaf(1.0f - gate, rv, bb[s]));
      sum = fmaf(val, ur[s], sum);
    }
    out[((size_t)b * L_ + i) * L_ + j] = sum;
  }
}

// ---------------------------------------------------------------------------
extern "C" void kernel_launch(void* const* d_in, const int* in_sizes, int n_in,
                              void* d_out, int out_size, void* d_ws, size_t ws_size,
                              hipStream_t stream) {
  const float* lstm1 = (const float*)d_in[0];
  const float* lstm2 = (const float*)d_in[1];
  const float* W     = (const float*)d_in[2];
  const float* V     = (const float*)d_in[3];
  const float* bias  = (const float*)d_in[4];
  const float* Wg    = (const float*)d_in[5];
  const float* bg    = (const float*)d_in[6];
  const float* u     = (const float*)d_in[7];

  // Workspace carve-up (~9.5 MB total).
  const size_t A_ELEMS   = (size_t)B_ * S_ * 32 * 2 * 32 * 16;  // 2,097,152 bf16
  const size_t Bp_ELEMS  = (size_t)B_ * 32 * 2 * 32 * 16;       //   262,144 bf16
  const size_t VEC_ELEMS = (size_t)B_ * L_ * S_;                //    32,768 f32
  char* ws = (char*)d_ws;
  __bf16* Ahi = (__bf16*)ws; ws += A_ELEMS * sizeof(__bf16);
  __bf16* Alo = (__bf16*)ws; ws += A_ELEMS * sizeof(__bf16);
  __bf16* Bhi = (__bf16*)ws; ws += Bp_ELEMS * sizeof(__bf16);
  __bf16* Blo = (__bf16*)ws; ws += Bp_ELEMS * sizeof(__bf16);
  float* v1 = (float*)ws; ws += VEC_ELEMS * sizeof(float);
  float* v2 = (float*)ws; ws += VEC_ELEMS * sizeof(float);
  float* g1 = (float*)ws; ws += VEC_ELEMS * sizeof(float);
  float* g2 = (float*)ws; ws += VEC_ELEMS * sizeof(float);

  grn_prep_A  <<<8192, 256, 0, stream>>>(lstm1, W, Ahi, Alo);
  grn_pack_B  <<<1024, 256, 0, stream>>>(lstm2, Bhi, Blo);
  grn_prep_vec<<<128,  256, 0, stream>>>(lstm1, lstm2, V, Wg, v1, v2, g1, g2);
  grn_main    <<<1024, 256, 0, stream>>>(Ahi, Alo, Bhi, Blo, v1, v2, g1, g2,
                                         bias, bg, u, (float*)d_out);
}
